// WalvePlainConvUNet_57423712748252
// MI455X (gfx1250) — compile-verified, matrix operand order checked by reading, not run on hardware
//
#include <hip/hip_runtime.h>

// ---------------------------------------------------------------------------
// Types for WMMA / TDM
// ---------------------------------------------------------------------------
typedef __attribute__((ext_vector_type(16))) _Float16 v16h;
typedef __attribute__((ext_vector_type(8)))  _Float16 v8h;
typedef __attribute__((ext_vector_type(8)))  float    v8f;
typedef __attribute__((ext_vector_type(4)))  unsigned int u32x4;
typedef __attribute__((ext_vector_type(4)))  int       i32x4;
typedef __attribute__((ext_vector_type(8)))  int       i32x8;

#define LDSPAD 40   // halves per LDS row (80B: 16B aligned, avoids bank conflicts)

union PkHalf { unsigned u; _Float16 h[2]; };

// ---------------------------------------------------------------------------
// Implicit-GEMM 3x3x3 conv (pad=1) using v_wmma_f32_16x16x32_f16.
//   GEMM view: M = Cout, N = Dout*Hout*Wout, K = Cin*27.
//   Block = 256 threads = 8 waves; block tile = 32 (Cout) x 128 (spatial);
//   wave grid 2(M) x 4(N); each wave computes 16x32 = two WMMA tiles per
//   K-step sharing one A fragment. LDS double-buffered, one barrier per step.
//   Staging is branch-free (clamped offsets + select) so loads batch.
//   Supports a second input pointer so decoder concat is fused.
// ---------------------------------------------------------------------------
__global__ __launch_bounds__(256)
void conv3d_wmma(const float* __restrict__ in0, int C0,
                 const float* __restrict__ in1, int C1,
                 const float* __restrict__ wgt,
                 const float* __restrict__ bias,
                 float* __restrict__ out,
                 int Cout, int Din, int Hin, int Win,
                 int Dout, int Hout, int Wout, int stride)
{
    __shared__ __align__(16) _Float16 lA[2][32][LDSPAD];    // weights [co][k]
    __shared__ __align__(16) _Float16 lB[2][128][LDSPAD];   // im2col  [sp][k]

    const int Cin  = C0 + C1;
    const int Ktot = Cin * 27;
    const int nSp  = Dout * Hout * Wout;
    const int n      = blockIdx.z;
    const int spBase = blockIdx.x * 128;
    const int coBase = blockIdx.y * 32;
    const int tid    = threadIdx.x;
    const int lane   = tid & 31;
    const int wave   = tid >> 5;
    const int coTile = wave >> 2;          // 0..1  (M tile)
    const int spTB   = (wave & 3) * 32;    // N base for this wave (two tiles)
    const int HW     = Hin * Win;

    // ---- branch-free A staging: 32 x 32 halves = 512 dword pairs ---------
    auto stageA = [&](int buf, int k0) {
#pragma unroll
        for (int t = 0; t < 2; ++t) {
            int idx = tid + t * 256;
            int row = idx >> 4;            // co_local 0..31
            int kp  = (idx & 15) * 2;      // even k within tile
            int co  = coBase + row;
            float v[2];
#pragma unroll
            for (int j = 0; j < 2; ++j) {
                int kg = k0 + kp + j;
                bool ok = (co < Cout) && (kg < Ktot);
                // OIDHW: flat = co*Cin*27 + ci*27 + r == co*Ktot + kg
                size_t off = ok ? ((size_t)co * Ktot + kg) : 0;
                float x = wgt[off];
                v[j] = ok ? x : 0.f;
            }
            PkHalf pk;
            pk.h[0] = (_Float16)v[0];
            pk.h[1] = (_Float16)v[1];
            *(unsigned*)&lA[buf][row][kp] = pk.u;
        }
    };

    // ---- branch-free B staging (im2col): 128 x 32 halves = 2048 pairs ----
    auto stageB = [&](int buf, int k0) {
#pragma unroll
        for (int t = 0; t < 8; ++t) {
            int idx   = tid + t * 256;
            int sp    = idx >> 4;          // 0..127
            int kp    = (idx & 15) * 2;
            int spIdx = spBase + sp;
            int ow = spIdx % Wout;
            int oh = (spIdx / Wout) % Hout;
            int od = spIdx / (Wout * Hout);
            float v[2];
#pragma unroll
            for (int j = 0; j < 2; ++j) {
                int kg = k0 + kp + j;
                int ci = kg / 27;
                int r  = kg - ci * 27;
                int kd = r / 9, kh = (r / 3) % 3, kw = r % 3;
                int id = od * stride + kd - 1;
                int ih = oh * stride + kh - 1;
                int iw = ow * stride + kw - 1;
                bool ok = (kg < Ktot) && (spIdx < nSp) &&
                          ((unsigned)id < (unsigned)Din) &&
                          ((unsigned)ih < (unsigned)Hin) &&
                          ((unsigned)iw < (unsigned)Win);
                bool second = ok && (ci >= C0);
                const float* base = second ? in1 : in0;
                int cEff = second ? (ci - C0) : ci;
                int Cb   = second ? C1 : C0;
                size_t off = (((size_t)n * Cb + cEff) * Din + id) * HW
                           + (size_t)ih * Win + iw;
                off = ok ? off : 0;
                float x = base[off];
                v[j] = ok ? x : 0.f;
            }
            PkHalf pk;
            pk.h[0] = (_Float16)v[0];
            pk.h[1] = (_Float16)v[1];
            *(unsigned*)&lB[buf][sp][kp] = pk.u;
        }
    };

    v8f acc0 = {}, acc1 = {};
    const int nSteps = (Ktot + 31) / 32;

    stageA(0, 0);
    stageB(0, 0);
    __syncthreads();

    for (int s = 0; s < nSteps; ++s) {
        const int cur = s & 1, nxt = cur ^ 1;

        // fragment loads per 16-bit A/B layout (2x ds_load_b128 each)
        const int row = coTile * 16 + (lane & 15);
        const int kb  = (lane >> 4) * 8;   // lanes 0-15: K 0..7/16..23

        v8h alo = *(const v8h*)&lA[cur][row][kb];
        v8h ahi = *(const v8h*)&lA[cur][row][kb + 16];
        v16h a  = __builtin_shufflevector(alo, ahi,
                    0,1,2,3,4,5,6,7,8,9,10,11,12,13,14,15);

        const int c0 = spTB + (lane & 15);
        v8h b0l = *(const v8h*)&lB[cur][c0][kb];
        v8h b0h = *(const v8h*)&lB[cur][c0][kb + 16];
        v16h b0 = __builtin_shufflevector(b0l, b0h,
                    0,1,2,3,4,5,6,7,8,9,10,11,12,13,14,15);
        v8h b1l = *(const v8h*)&lB[cur][c0 + 16][kb];
        v8h b1h = *(const v8h*)&lB[cur][c0 + 16][kb + 16];
        v16h b1 = __builtin_shufflevector(b1l, b1h,
                    0,1,2,3,4,5,6,7,8,9,10,11,12,13,14,15);

        acc0 = __builtin_amdgcn_wmma_f32_16x16x32_f16(
                   false, a, false, b0, (short)0, acc0, false, false);
        acc1 = __builtin_amdgcn_wmma_f32_16x16x32_f16(
                   false, a, false, b1, (short)0, acc1, false, false);

        if (s + 1 < nSteps) {
            if (tid == 0)
                __builtin_prefetch(&wgt[(size_t)coBase * Ktot + (s + 1) * 32], 0, 1);
            stageA(nxt, (s + 1) * 32);     // overlaps WMMA; other buffer
            stageB(nxt, (s + 1) * 32);
        }
        __syncthreads();
    }

    // ---- epilogue: C/D layout (VGPR r: lanes0-15 M=r, lanes16-31 M=r+8) --
    const int mBase = (lane >> 4) * 8;
    const int nIdx  = lane & 15;
    const int sp0   = spBase + spTB + nIdx;
    const int sp1   = sp0 + 16;
#pragma unroll
    for (int r = 0; r < 8; ++r) {
        int co = coBase + coTile * 16 + r + mBase;
        if (co < Cout) {
            float bv = bias ? bias[co] : 0.f;
            size_t rowOff = ((size_t)n * Cout + co) * nSp;
            if (sp0 < nSp) out[rowOff + sp0] = acc0[r] + bv;
            if (sp1 < nSp) out[rowOff + sp1] = acc1[r] + bv;
        }
    }
}

// ---------------------------------------------------------------------------
// Fused InstanceNorm + LeakyReLU (in place). One block per (c, n).
// ---------------------------------------------------------------------------
__global__ __launch_bounds__(256)
void inorm_lrelu(float* __restrict__ x, const float* __restrict__ g,
                 const float* __restrict__ b, int C, int nSp)
{
    const int c = blockIdx.x, n = blockIdx.y;
    float* p = x + ((size_t)n * C + c) * nSp;
    float s = 0.f, s2 = 0.f;
    for (int i = threadIdx.x; i < nSp; i += 256) {
        float v = p[i]; s += v; s2 += v * v;
    }
    __shared__ float rs[256], rq[256];
    rs[threadIdx.x] = s; rq[threadIdx.x] = s2;
    __syncthreads();
    for (int o = 128; o > 0; o >>= 1) {
        if (threadIdx.x < o) {
            rs[threadIdx.x] += rs[threadIdx.x + o];
            rq[threadIdx.x] += rq[threadIdx.x + o];
        }
        __syncthreads();
    }
    const float m   = rs[0] / (float)nSp;
    const float var = rq[0] / (float)nSp - m * m;
    const float inv = rsqrtf(var + 1e-5f);
    const float gg = g[c], bb = b[c];
    for (int i = threadIdx.x; i < nSp; i += 256) {
        float v = (p[i] - m) * inv * gg + bb;
        p[i] = v >= 0.f ? v : 0.01f * v;
    }
}

// ---------------------------------------------------------------------------
// Haar-LL + maxpool 2x2x2, concatenated along channels: out[N][2C][D/2]^3
// ---------------------------------------------------------------------------
__global__ void haar_pool_cat(const float* __restrict__ in, float* __restrict__ out,
                              int N, int C, int D, int H, int W)
{
    const int Dp = D / 2, Hp = H / 2, Wp = W / 2;
    const size_t pooled = (size_t)Dp * Hp * Wp;
    const size_t total  = (size_t)N * C * pooled;
    const size_t gs = (size_t)gridDim.x * blockDim.x;
    for (size_t i = (size_t)blockIdx.x * blockDim.x + threadIdx.x; i < total; i += gs) {
        int wv = (int)(i % Wp);
        int h  = (int)((i / Wp) % Hp);
        int d  = (int)((i / ((size_t)Wp * Hp)) % Dp);
        int c  = (int)((i / pooled) % C);
        int n  = (int)(i / (pooled * C));
        const float* p = in + (((size_t)n * C + c) * D + 2 * d) * H * W;
        float s = 0.f, mx = -3.4e38f;
#pragma unroll
        for (int a = 0; a < 2; ++a)
#pragma unroll
            for (int bb = 0; bb < 2; ++bb)
#pragma unroll
                for (int cc = 0; cc < 2; ++cc) {
                    float v = p[a * H * W + (2 * h + bb) * W + (2 * wv + cc)];
                    s += v; mx = fmaxf(mx, v);
                }
        size_t sp = ((size_t)d * Hp + h) * Wp + wv;
        out[((size_t)(n * 2 * C + c)) * pooled + sp]     = s * 0.125f * 2.8284271247461903f;
        out[((size_t)(n * 2 * C + c + C)) * pooled + sp] = mx;
    }
}

__global__ void maxpool_k(const float* __restrict__ in, float* __restrict__ out,
                          int N, int C, int D, int H, int W)
{
    const int Dp = D / 2, Hp = H / 2, Wp = W / 2;
    const size_t pooled = (size_t)Dp * Hp * Wp;
    const size_t total  = (size_t)N * C * pooled;
    const size_t gs = (size_t)gridDim.x * blockDim.x;
    for (size_t i = (size_t)blockIdx.x * blockDim.x + threadIdx.x; i < total; i += gs) {
        int wv = (int)(i % Wp);
        int h  = (int)((i / Wp) % Hp);
        int d  = (int)((i / ((size_t)Wp * Hp)) % Dp);
        int c  = (int)((i / pooled) % C);
        int n  = (int)(i / (pooled * C));
        const float* p = in + (((size_t)n * C + c) * D + 2 * d) * H * W;
        float mx = -3.4e38f;
#pragma unroll
        for (int a = 0; a < 2; ++a)
#pragma unroll
            for (int bb = 0; bb < 2; ++bb)
#pragma unroll
                for (int cc = 0; cc < 2; ++cc)
                    mx = fmaxf(mx, p[a * H * W + (2 * h + bb) * W + (2 * wv + cc)]);
        out[i] = mx;
    }
}

// ---------------------------------------------------------------------------
// ConvTranspose3d kernel=stride=2, no bias. w: [Cin][Cout][2][2][2]
// ---------------------------------------------------------------------------
__global__ void upconv2_k(const float* __restrict__ in, const float* __restrict__ w,
                          float* __restrict__ out, int N, int Cin, int Cout,
                          int D, int H, int W)
{
    const int D2 = 2 * D, H2 = 2 * H, W2 = 2 * W;
    const size_t total = (size_t)N * Cout * D2 * H2 * W2;
    const size_t gs = (size_t)gridDim.x * blockDim.x;
    for (size_t t = (size_t)blockIdx.x * blockDim.x + threadIdx.x; t < total; t += gs) {
        int ow = (int)(t % W2);
        int oh = (int)((t / W2) % H2);
        int od = (int)((t / ((size_t)W2 * H2)) % D2);
        int o  = (int)((t / ((size_t)W2 * H2 * D2)) % Cout);
        int n  = (int)(t / ((size_t)W2 * H2 * D2 * Cout));
        int d = od >> 1, i = od & 1;
        int h = oh >> 1, j = oh & 1;
        int wv = ow >> 1, k = ow & 1;
        float acc = 0.f;
        for (int c = 0; c < Cin; ++c)
            acc += in[(((size_t)n * Cin + c) * D + d) * H * W + h * W + wv] *
                   w[((((size_t)c * Cout + o) * 2 + i) * 2 + j) * 2 + k];
        out[t] = acc;
    }
}

// ---------------------------------------------------------------------------
// 1x1x1 seg head
// ---------------------------------------------------------------------------
__global__ void seg_conv1(const float* __restrict__ in, const float* __restrict__ w,
                          const float* __restrict__ b, float* __restrict__ out,
                          int Cin, int K, int nSp, int N)
{
    const size_t total = (size_t)N * K * nSp;
    const size_t gs = (size_t)gridDim.x * blockDim.x;
    for (size_t i = (size_t)blockIdx.x * blockDim.x + threadIdx.x; i < total; i += gs) {
        int sp = (int)(i % nSp);
        int k  = (int)((i / nSp) % K);
        int n  = (int)(i / ((size_t)nSp * K));
        float acc = b[k];
        for (int c = 0; c < Cin; ++c)
            acc += w[k * Cin + c] * in[((size_t)n * Cin + c) * nSp + sp];
        out[i] = acc;
    }
}

// ---------------------------------------------------------------------------
// 2x trilinear upsample, align_corners=False
// ---------------------------------------------------------------------------
__device__ inline void lin_idx(int o, int D, int& i0, int& i1, float& t)
{
    float pos = (o + 0.5f) * 0.5f - 0.5f;
    float f = floorf(pos);
    t = pos - f;
    int i = (int)f;
    i0 = min(max(i, 0), D - 1);
    i1 = min(max(i + 1, 0), D - 1);
}

__global__ void trilinear2(const float* __restrict__ in, float* __restrict__ out,
                           int N, int C, int D, int H, int W)
{
    const int D2 = 2 * D, H2 = 2 * H, W2 = 2 * W;
    const size_t total = (size_t)N * C * D2 * H2 * W2;
    const size_t gs = (size_t)gridDim.x * blockDim.x;
    for (size_t t = (size_t)blockIdx.x * blockDim.x + threadIdx.x; t < total; t += gs) {
        int ow = (int)(t % W2);
        int oh = (int)((t / W2) % H2);
        int od = (int)((t / ((size_t)W2 * H2)) % D2);
        int c  = (int)((t / ((size_t)W2 * H2 * D2)) % C);
        int n  = (int)(t / ((size_t)W2 * H2 * D2 * C));
        int d0, d1, h0, h1, w0, w1; float td, th, tw;
        lin_idx(od, D, d0, d1, td);
        lin_idx(oh, H, h0, h1, th);
        lin_idx(ow, W, w0, w1, tw);
        const float* p = in + ((size_t)n * C + c) * D * H * W;
        auto at = [&](int d, int h, int w_) { return p[((size_t)d * H + h) * W + w_]; };
        float v =
            (1-td)*((1-th)*((1-tw)*at(d0,h0,w0) + tw*at(d0,h0,w1)) +
                       th *((1-tw)*at(d0,h1,w0) + tw*at(d0,h1,w1))) +
               td *((1-th)*((1-tw)*at(d1,h0,w0) + tw*at(d1,h0,w1)) +
                       th *((1-tw)*at(d1,h1,w0) + tw*at(d1,h1,w1)));
        out[t] = v;
    }
}

// ---------------------------------------------------------------------------
// Tensor Data Mover warm-up: issue a real tensor_load_to_lds + s_wait_tensorcnt
// ---------------------------------------------------------------------------
__global__ void tdm_prefetch(const float* __restrict__ src, float* __restrict__ sink)
{
    __shared__ __align__(16) float stage[1024];
    unsigned long long ga = (unsigned long long)src;
    unsigned lds = (unsigned)(unsigned long long)(void*)stage;

    u32x4 g0;
    g0[0] = 1u;                                   // count = 1 (valid descriptor)
    g0[1] = lds;                                  // lds_addr
    g0[2] = (unsigned)(ga & 0xffffffffu);         // global_addr lo
    g0[3] = (unsigned)((ga >> 32) & 0x01ffffffu) | (2u << 30);  // hi + type=2
    i32x8 g1 = {};
    g1[0] = (2 << 16);                            // data_size = 4B, no multicast
    g1[1] = 0;
    g1[2] = 1024 << 16;                           // tensor_dim0 (spans bits 79:48)
    g1[3] = (1 << 16) | 1024;                     // tile_dim0 = 1024, tile_dim1 = 1
    i32x4 g2 = {};
    i32x4 g3 = {};
    if (threadIdx.x == 0) {
#if __clang_major__ >= 23
        i32x8 g4 = {};
        __builtin_amdgcn_tensor_load_to_lds(g0, g1, g2, g3, g4, 0);
#else
        __builtin_amdgcn_tensor_load_to_lds(g0, g1, g2, g3, 0);
#endif
    }
    __builtin_amdgcn_s_wait_tensorcnt(0);
    __syncthreads();
    sink[threadIdx.x] = stage[threadIdx.x];
}

// ---------------------------------------------------------------------------
// Host orchestration
// ---------------------------------------------------------------------------
extern "C" void kernel_launch(void* const* d_in, const int* in_sizes, int n_in,
                              void* d_out, int out_size, void* d_ws, size_t ws_size,
                              hipStream_t stream)
{
    (void)in_sizes; (void)n_in; (void)out_size; (void)ws_size;
    const int N = 2;
    static const int FEATS[6]   = {32, 64, 125, 250, 320, 320};
    static const int STRIDES[6] = {1, 2, 2, 2, 2, 2};

    // pytree flatten order (sorted dict keys):
    //  dec[0..4]: convs0.{b,g,w}, convs1.{b,g,w}, up_w     (7 each -> 0..34)
    //  enc[0..5]: conv0.{b,g,w}, conv1.{b,g,w}             (6 each -> 35..70)
    //  seg: b, w                                           (71, 72)
    //  wconv[0..2]: b, w                                   (73..78)
    //  x                                                   (79)
    auto F = [&](int i) { return (const float*)d_in[i]; };
    auto decB = [&](int j, int c) { return F(j * 7 + c * 3 + 0); };
    auto decG = [&](int j, int c) { return F(j * 7 + c * 3 + 1); };
    auto decW = [&](int j, int c) { return F(j * 7 + c * 3 + 2); };
    auto decU = [&](int j)        { return F(j * 7 + 6); };
    auto encB = [&](int s, int c) { return F(35 + s * 6 + c * 3 + 0); };
    auto encG = [&](int s, int c) { return F(35 + s * 6 + c * 3 + 1); };
    auto encW = [&](int s, int c) { return F(35 + s * 6 + c * 3 + 2); };
    const float* segB = F(71);
    const float* segW = F(72);
    auto wcB = [&](int s) { return F(73 + s * 2 + 0); };
    auto wcW = [&](int s) { return F(73 + s * 2 + 1); };
    const float* x = F(79);

    // bump allocator over d_ws
    char* wsp = (char*)d_ws;
    size_t off = 0;
    auto alloc = [&](size_t elems) -> float* {
        float* p = (float*)(wsp + off);
        off += ((elems * sizeof(float) + 255) & ~(size_t)255);
        return p;
    };

    auto conv = [&](const float* in0, int C0, const float* in1, int C1,
                    const float* w, const float* b, float* out,
                    int Cout, int Di, int Do, int stride) {
        int nSp = Do * Do * Do;
        dim3 g((nSp + 127) / 128, (Cout + 31) / 32, N);
        conv3d_wmma<<<g, dim3(256), 0, stream>>>(in0, C0, in1, C1, w, b, out,
                                                 Cout, Di, Di, Di, Do, Do, Do, stride);
    };
    auto inorm = [&](float* p, const float* g, const float* b, int C, int nSp) {
        inorm_lrelu<<<dim3(C, N), dim3(256), 0, stream>>>(p, g, b, C, nSp);
    };
    auto gsz = [](size_t total) { return dim3((unsigned)((total + 255) / 256)); };

    // TDM warm-up of the first-layer weights
    float* tdm_sink = alloc(1024);
    tdm_prefetch<<<dim3(1), dim3(32), 0, stream>>>(encW(0, 0), tdm_sink);

    // ---------------- encoder ----------------
    float* skip[6]; int dim[6];
    const float* cur = x;
    int prevC = 1, prevD = 64;
    for (int s = 0; s < 6; ++s) {
        int st = STRIDES[s];
        int dOut = prevD / st;
        int o = FEATS[s];
        int nSp = dOut * dOut * dOut;
        float* a = alloc((size_t)N * o * nSp);
        conv(cur, prevC, nullptr, 0, encW(s, 0), nullptr, a, o, prevD, dOut, st);
        inorm(a, encG(s, 0), encB(s, 0), o, nSp);
        float* b2 = alloc((size_t)N * o * nSp);
        conv(a, o, nullptr, 0, encW(s, 1), nullptr, b2, o, dOut, dOut, 1);
        inorm(b2, encG(s, 1), encB(s, 1), o, nSp);
        skip[s] = b2; dim[s] = dOut;
        cur = b2; prevC = o; prevD = dOut;
    }

    // ---------------- wavelet-enhanced skips ----------------
    float* enh[6]; int edim[6];
    for (int s = 0; s < 6; ++s) {
        int C = FEATS[s];
        int Dp = dim[s] / 2;
        size_t pooled = (size_t)Dp * Dp * Dp;
        size_t total = (size_t)N * C * pooled;
        if (s < 3) {
            float* cat = alloc((size_t)N * 2 * C * pooled);
            haar_pool_cat<<<gsz(total), dim3(256), 0, stream>>>(
                skip[s], cat, N, C, dim[s], dim[s], dim[s]);
            float* eo = alloc(total);
            conv(cat, 2 * C, nullptr, 0, wcW(s), wcB(s), eo, C, Dp, Dp, 1);
            enh[s] = eo;
        } else {
            float* eo = alloc(total);
            maxpool_k<<<gsz(total), dim3(256), 0, stream>>>(
                skip[s], eo, N, C, dim[s], dim[s], dim[s]);
            enh[s] = eo;
        }
        edim[s] = Dp;
    }

    // ---------------- decoder ----------------
    float* h = enh[5];
    int hC = FEATS[5], hD = edim[5];     // 320 @ 1^3
    for (int j = 0; j < 5; ++j) {
        int sf = FEATS[4 - j];
        int upD = hD * 2;
        size_t nSp = (size_t)upD * upD * upD;
        float* up = alloc((size_t)N * sf * nSp);
        upconv2_k<<<gsz((size_t)N * sf * nSp), dim3(256), 0, stream>>>(
            h, decU(j), up, N, hC, sf, hD, hD, hD);
        // concat(up, enh[4-j]) fused into conv via dual-pointer input
        const float* e = enh[4 - j];
        int eC = FEATS[4 - j];
        float* c1 = alloc((size_t)N * sf * nSp);
        conv(up, sf, e, eC, decW(j, 0), nullptr, c1, sf, upD, upD, 1);
        inorm(c1, decG(j, 0), decB(j, 0), sf, (int)nSp);
        float* c2 = alloc((size_t)N * sf * nSp);
        conv(c1, sf, nullptr, 0, decW(j, 1), nullptr, c2, sf, upD, upD, 1);
        inorm(c2, decG(j, 1), decB(j, 1), sf, (int)nSp);
        h = c2; hC = sf; hD = upD;
    }

    // ---------------- seg head + 2x trilinear ----------------
    int nSpSeg = hD * hD * hD;
    float* seg = alloc((size_t)N * 3 * nSpSeg);
    seg_conv1<<<gsz((size_t)N * 3 * nSpSeg), dim3(256), 0, stream>>>(
        h, segW, segB, seg, hC, 3, nSpSeg, N);
    trilinear2<<<gsz((size_t)N * 3 * nSpSeg * 8), dim3(256), 0, stream>>>(
        seg, (float*)d_out, N, 3, hD, hD, hD);
}
